// AxialBottleneck_54795192763021
// MI455X (gfx1250) — matrix operand order repaired and checked by reading, not compiled, not run on hardware
//
#include <hip/hip_runtime.h>

#define EPS    1e-5f
#define P_TOT  32768   // 256*128 spatial positions
#define NBLK   128     // GEMM N-tile per block
#define LDA    72      // padded LDS row stride (halfs): 144B, 16B-aligned, bank-skewed

typedef __attribute__((ext_vector_type(16))) _Float16 v16h;
typedef __attribute__((ext_vector_type(8)))  _Float16 h8;
typedef __attribute__((ext_vector_type(8)))  float    v8f;

union FragH { v16h v; h8 h[2]; _Float16 e[16]; };
union FragF { v8f  v; float e[8]; };

// ---------------------------------------------------------------------------
// Fused 1x1-conv GEMM + BatchNorm + epilogue (templated, branch-free):
//   Out[m, p] = epi( scale[m] * sum_k W[m,k]*In[k,p] + bias[m] (+ resid[m,p]) )
// K fixed at 64; M in {64,128} (compile-time -> fully unrolled row tiles).
// EPI: 0 = none, 1 = relu, 2 = residual + relu.
// Core: v_wmma_f32_16x16x32_f16, f32 accumulate.  Both operand tiles are laid
// out in LDS so every lane's fragment is 16B-contiguous chunks -> pure
// ds_load_b128 fragment fetches:
//   sA  : row-major  M x 64   (A 16x32 layout: lane reads K-runs of 8)
//   sBT : col-major  128 x 64 (B 32x16 layout: lane reads K-runs of 8)
// Block = 256 threads = 8 waves; each wave owns one 16-column strip; its two
// B fragments (kk=0,32) are loaded once and reused across all M/16 row tiles.
// ---------------------------------------------------------------------------
template<int M, int EPI>
__global__ void gemm_bn_epi(const float* __restrict__ W,
                            const float* __restrict__ In,
                            const float* __restrict__ bn,
                            const float* __restrict__ resid,
                            float* __restrict__ Out)
{
    __shared__ _Float16 sA [M * LDA];      // weights, row-major   [m][k]
    __shared__ _Float16 sBT[NBLK * LDA];   // activ., transposed   [n][k]
    __shared__ float sScale[M], sBias[M];

    const int tid = threadIdx.x;
    const int n0  = blockIdx.x * NBLK;

    // stage A (weights): float4 global loads, contiguous f16 row writes
    #pragma unroll
    for (int i = tid; i < M * 16; i += 256) {
        int m = i >> 4, k4 = (i & 15) * 4;
        const float4 f = *(const float4*)(W + m * 64 + k4);
        _Float16* d = &sA[m * LDA + k4];
        d[0] = (_Float16)f.x; d[1] = (_Float16)f.y;
        d[2] = (_Float16)f.z; d[3] = (_Float16)f.w;
    }
    // stage B (activations): float4 (b128) coalesced global loads along n,
    // transposed scatter into LDS so K is contiguous per column
    #pragma unroll
    for (int i = tid; i < 64 * (NBLK / 4); i += 256) {
        int k = i >> 5, n4 = (i & 31) * 4;
        const float4 f = *(const float4*)(In + k * P_TOT + n0 + n4);
        sBT[(n4 + 0) * LDA + k] = (_Float16)f.x;
        sBT[(n4 + 1) * LDA + k] = (_Float16)f.y;
        sBT[(n4 + 2) * LDA + k] = (_Float16)f.z;
        sBT[(n4 + 3) * LDA + k] = (_Float16)f.w;
    }
    // fold BN params: scale = g/sqrt(v+eps), bias = b - m*scale
    if (tid < M) {
        float g  = bn[tid];
        float b  = bn[M + tid];
        float mu = bn[2 * M + tid];
        float vv = bn[3 * M + tid];
        float s  = g * rsqrtf(vv + EPS);
        sScale[tid] = s;
        sBias[tid]  = b - mu * s;
    }
    __syncthreads();

    const int wave = tid >> 5;
    const int lane = tid & 31;
    const int hs   = lane >> 4;        // half-select (lanes 16..31)
    const int l16  = lane & 15;
    const int col  = n0 + wave * 16 + l16;

    // B fragments for this wave's column strip (loop-invariant across row
    // tiles).  ISA 16-bit B 32x16 layout: e[0..7] <-> K = 8*hs.., e[8..15] <->
    // K = 16+8*hs.., N = lane%16 -> two 8-half runs = two ds_load_b128.
    const int nRow = (wave * 16 + l16) * LDA;
    FragH b0, b1;
    b0.h[0] = *(const h8*)&sBT[nRow + 8 * hs];
    b0.h[1] = *(const h8*)&sBT[nRow + 16 + 8 * hs];
    b1.h[0] = *(const h8*)&sBT[nRow + 32 + 8 * hs];
    b1.h[1] = *(const h8*)&sBT[nRow + 48 + 8 * hs];

    #pragma unroll
    for (int rt = 0; rt < (M >> 4); ++rt) {
        FragF acc;
        #pragma unroll
        for (int r = 0; r < 8; ++r) acc.e[r] = 0.f;

        // ISA 16-bit A 16x32 layout: e[0..7] <-> K = 8*hs..8*hs+7,
        // e[8..15] <-> K = 16+8*hs.. -> 8-half contiguous runs = ds_load_b128
        const int mRow = (rt * 16 + l16) * LDA;
        FragH a0, a1;
        a0.h[0] = *(const h8*)&sA[mRow + 8 * hs];
        a0.h[1] = *(const h8*)&sA[mRow + 16 + 8 * hs];
        a1.h[0] = *(const h8*)&sA[mRow + 32 + 8 * hs];
        a1.h[1] = *(const h8*)&sA[mRow + 48 + 8 * hs];

        acc.v = __builtin_amdgcn_wmma_f32_16x16x32_f16(
                    false, a0.v, false, b0.v, (short)0, acc.v, false, false);
        acc.v = __builtin_amdgcn_wmma_f32_16x16x32_f16(
                    false, a1.v, false, b1.v, (short)0, acc.v, false, false);

        #pragma unroll
        for (int r = 0; r < 8; ++r) {
            int mr  = rt * 16 + r + 8 * hs;     // C/D layout: row = r + 8*hs
            float v = sScale[mr] * acc.e[r] + sBias[mr];
            if (EPI == 2) v += resid[mr * P_TOT + col];
            if (EPI != 0) v = fmaxf(v, 0.f);
            Out[mr * P_TOT + col] = v;
        }
    }
}

// ---------------------------------------------------------------------------
// Axial attention core for one (sequence, head):
//   logits[x,y] = s_qk*(q.k) + s_qr*(q.q_enc[x,y]) + s_kr*(k.k_enc[y,x]) + biases
//   out[d,x]    = softmax_y(logits)[x,y] . (v[d,y] + v_enc[d,x,y])
// kqv channels per head: [0..3]=k, [4..7]=q, [8..15]=v (reference split order).
// rel is (16, 2L-1); rel[d, y-x+L-1] is the (x,y) relative encoding.
// VALU by design: the contractions have K=4/8 (WMMA 16x16x32 would waste 8x
// of K) and the rel-enc gathers + softmax dominate.  L is compile-time so the
// per-lane y loops (L/32 iterations) fully unroll.  Wave32 shfl reductions.
// ---------------------------------------------------------------------------
template<int L, int DORELU>
__global__ void axial_attn(const float* __restrict__ kqv,
                           const float* __restrict__ rel,
                           const float* __restrict__ lbn,
                           float* __restrict__ out,
                           int stride, int baseMul)
{
    __shared__ float sm[16 * L];       // k(4L) | q(4L) | v(8L)
    const int head = blockIdx.x & 7;
    const int seq  = blockIdx.x >> 3;
    const int base = seq * baseMul;
    const int RW   = 2 * L - 1;

    for (int i = threadIdx.x; i < 16 * L; i += 256) {
        int c = i / L, t = i - c * L;
        sm[c * L + t] = kqv[(head * 16 + c) * P_TOT + base + t * stride];
    }
    __syncthreads();

    const float* sk = sm;            // k
    const float* sq = sm + 4 * L;    // q
    const float* sv = sm + 8 * L;    // v

    // fold logits-BN: 24 channels = [qk(8) | qr(8) | kr(8)], summed over groups
    float sqk, bqk, sqr, bqr, skr, bkr;
    {
        int c0 = head, c1 = 8 + head, c2 = 16 + head;
        sqk = lbn[c0] * rsqrtf(lbn[72 + c0] + EPS); bqk = lbn[24 + c0] - lbn[48 + c0] * sqk;
        sqr = lbn[c1] * rsqrtf(lbn[72 + c1] + EPS); bqr = lbn[24 + c1] - lbn[48 + c1] * sqr;
        skr = lbn[c2] * rsqrtf(lbn[72 + c2] + EPS); bkr = lbn[24 + c2] - lbn[48 + c2] * skr;
    }

    const int wave = threadIdx.x >> 5;
    const int lane = threadIdx.x & 31;
    constexpr int NJ = L >> 5;         // y-values per lane (8 for L=256, 4 for 128)

    for (int x = wave; x < L; x += 8) {
        const float qx0 = sq[0 * L + x], qx1 = sq[1 * L + x];
        const float qx2 = sq[2 * L + x], qx3 = sq[3 * L + x];
        float lv[NJ];
        float mx = -3.0e38f;
        #pragma unroll
        for (int j = 0; j < NJ; ++j) {
            int y  = (j << 5) + lane;
            int rq = y - x + (L - 1);
            int rk = x - y + (L - 1);
            float k0 = sk[0 * L + y], k1 = sk[1 * L + y];
            float k2 = sk[2 * L + y], k3 = sk[3 * L + y];
            float qk = qx0 * k0 + qx1 * k1 + qx2 * k2 + qx3 * k3;
            float qr = qx0 * rel[0 * RW + rq] + qx1 * rel[1 * RW + rq]
                     + qx2 * rel[2 * RW + rq] + qx3 * rel[3 * RW + rq];
            float kr = k0 * rel[4 * RW + rk] + k1 * rel[5 * RW + rk]
                     + k2 * rel[6 * RW + rk] + k3 * rel[7 * RW + rk];
            float l = sqk * qk + bqk + sqr * qr + bqr + skr * kr + bkr;
            lv[j] = l;
            mx = fmaxf(mx, l);
        }
        #pragma unroll
        for (int o = 16; o > 0; o >>= 1) mx = fmaxf(mx, __shfl_xor(mx, o, 32));
        float sum = 0.f;
        #pragma unroll
        for (int j = 0; j < NJ; ++j) { lv[j] = __expf(lv[j] - mx); sum += lv[j]; }
        #pragma unroll
        for (int o = 16; o > 0; o >>= 1) sum += __shfl_xor(sum, o, 32);
        const float inv = 1.0f / sum;

        #pragma unroll
        for (int d = 0; d < 8; ++d) {
            float acc = 0.f;
            #pragma unroll
            for (int j = 0; j < NJ; ++j) {
                int y = (j << 5) + lane;
                acc += lv[j] * (sv[d * L + y] + rel[(8 + d) * RW + (y - x + (L - 1))]);
            }
            #pragma unroll
            for (int o = 16; o > 0; o >>= 1) acc += __shfl_xor(acc, o, 32);
            if (lane == 0) {
                float v = acc * inv;
                if (DORELU) v = fmaxf(v, 0.f);
                out[(head * 8 + d) * P_TOT + base + x * stride] = v;
            }
        }
    }
}

extern "C" void kernel_launch(void* const* d_in, const int* in_sizes, int n_in,
                              void* d_out, int out_size, void* d_ws, size_t ws_size,
                              hipStream_t stream) {
    const float* x           = (const float*)d_in[0];   // (1,64,256,128)
    const float* conv1_w     = (const float*)d_in[1];   // (64,64)
    const float* bn1         = (const float*)d_in[2];   // (4,64)
    const float* kqv_w_h     = (const float*)d_in[3];   // (128,64)
    const float* kqv_bn_h    = (const float*)d_in[4];   // (4,128)
    const float* logits_bn_h = (const float*)d_in[5];   // (4,24)
    const float* rel_enc_h   = (const float*)d_in[6];   // (16,511)
    const float* kqv_w_w     = (const float*)d_in[7];   // (128,64)
    const float* kqv_bn_w    = (const float*)d_in[8];   // (4,128)
    const float* logits_bn_w = (const float*)d_in[9];   // (4,24)
    const float* rel_enc_w   = (const float*)d_in[10];  // (16,255)
    const float* conv3_w     = (const float*)d_in[11];  // (64,64)
    const float* bn3         = (const float*)d_in[12];  // (4,64)
    float* out = (float*)d_out;
    float* ws  = (float*)d_ws;

    const int P = P_TOT;
    float* buf1 = ws;             // 64*P  : conv1 output (bn+relu)
    float* kqvb = ws + 64 * P;    // 128*P : kqv projection (reused H then W)
    float* buf2 = ws + 192 * P;   // 64*P  : axial-H output
    float* buf3 = ws + 256 * P;   // 64*P  : relu(axial-W output)

    // 1. conv1 + bn1 + relu
    gemm_bn_epi<64, 1><<<P / NBLK, 256, 0, stream>>>(conv1_w, x, bn1, nullptr, buf1);
    // 2. kqv projection (H) + bn
    gemm_bn_epi<128, 0><<<P / NBLK, 256, 0, stream>>>(kqv_w_h, buf1, kqv_bn_h, nullptr, kqvb);
    // 3. axial attention along H: 128 sequences (one per w) x 8 heads, L=256
    axial_attn<256, 0><<<128 * 8, 256, 0, stream>>>(kqvb, rel_enc_h, logits_bn_h, buf2,
                                                    /*stride=*/128, /*baseMul=*/1);
    // 4. kqv projection (W) + bn
    gemm_bn_epi<128, 0><<<P / NBLK, 256, 0, stream>>>(kqv_w_w, buf2, kqv_bn_w, nullptr, kqvb);
    // 5. axial attention along W: 256 sequences (one per h) x 8 heads, L=128; relu epilogue
    axial_attn<128, 1><<<256 * 8, 256, 0, stream>>>(kqvb, rel_enc_w, logits_bn_w, buf3,
                                                    /*stride=*/1, /*baseMul=*/128);
    // 6. conv3 + bn3 + residual(x) + relu
    gemm_bn_epi<64, 2><<<P / NBLK, 256, 0, stream>>>(conv3_w, buf3, bn3, x, out);
}
